// ConfidenceAwareGovernor_72653666779171
// MI455X (gfx1250) — compile-verified, matrix operand order checked by reading, not run on hardware
//
#include <hip/hip_runtime.h>
#include <hip/hip_bf16.h>

typedef float v2f  __attribute__((ext_vector_type(2)));
typedef float v4f  __attribute__((ext_vector_type(4)));
typedef float v8f  __attribute__((ext_vector_type(8)));
typedef int   v4i  __attribute__((ext_vector_type(4)));
typedef _Float16 v16h __attribute__((ext_vector_type(16)));

#define B_   32
#define T_   4096
#define C_   256
#define NPS  (T_ * C_)           // 1,048,576 elements per sample
#define NBIN 8192                // top-13-bit radix bins of |x| pattern
#define BINSHIFT 18              // (bits & 0x7fffffff) >> 18 -> 13 bits
#define CAP  32768               // candidate capacity per sample

// workspace layout (bytes)
#define OFF_DSUM   0
#define OFF_COUNT  128
#define OFF_BINLO  256
#define OFF_BINHI  384
#define OFF_R0     512
#define OFF_FRAC   640
#define OFF_THRESH 768
#define OFF_HIST   1024
#define OFF_CAND   (OFF_HIST + B_ * NBIN * 4)           // 1,049,600
#define ZERO_BYTES OFF_CAND                              // zero dsum..hist

#if defined(__has_builtin)
#  if __has_builtin(__builtin_amdgcn_wmma_f32_16x16x4_f32)
#    define WMMA_PATH 1
#  elif __has_builtin(__builtin_amdgcn_wmma_f32_16x16x32_f16)
#    define WMMA_PATH 2
#  else
#    define WMMA_PATH 0
#  endif
#  if __has_builtin(__builtin_amdgcn_global_load_async_to_lds_b128)
#    define HAVE_ASYNC_LDS 1
#  else
#    define HAVE_ASYNC_LDS 0
#  endif
#else
#  define WMMA_PATH 0
#  define HAVE_ASYNC_LDS 0
#endif

#if HAVE_ASYNC_LDS
#  define AS_GLOBAL __attribute__((address_space(1)))
#  define AS_LDS    __attribute__((address_space(3)))
#endif

// ---------------------------------------------------------------------------
// Pass A: fused (a) per-token MSE -> uncertainty -> per-sample sum (WMMA f32
// 16x16x4 reduction over C), (b) 13-bit radix histogram of |s| in LDS.
// Teacher tiles are staged memory->LDS with GLOBAL_LOAD_ASYNC_TO_LDS_B128
// (CPol TH=NT so the one-shot teacher stream does not evict the student
// tensor from L2); the wave histograms the student chunk while the async
// copies are in flight, then s_wait_asynccnt 0 and diffs from LDS.
// 512 WGs (16 per sample), 256 threads, each wave handles 2 groups of 16 tokens.
// ---------------------------------------------------------------------------
__global__ __launch_bounds__(256) void k_pass_a(const float* __restrict__ s,
                                                const float* __restrict__ t,
                                                float* __restrict__ dsum,
                                                unsigned* __restrict__ ghist) {
  __shared__ unsigned lhist[NBIN];           // 32 KB
#if HAVE_ASYNC_LDS
  __shared__ float tstage[8 * 2048];         // 64 KB: 8 KB per wave, reused
#endif
  const int tid  = threadIdx.x;
  const int b    = blockIdx.x >> 4;            // 16 WGs per sample
  const int wgTok = (blockIdx.x & 15) * 256;   // 256 tokens per WG
  for (int i = tid; i < NBIN; i += 256) lhist[i] = 0u;
  __syncthreads();

  const int lane = tid & 31;
  const int wave = tid >> 5;
  float waveAcc = 0.0f;
  float dupInv  = 1.0f / 16.0f;   // WMMA D rows are replicated 16x across lanes

  for (int it = 0; it < 2; ++it) {
    const int tokBase = wgTok + (wave * 2 + it) * 16;
    const int tok     = tokBase + (lane & 15);
    const int kslot   = (lane >> 4) << 1;      // K = {0,1} or {2,3}
    const float* sp = s + ((size_t)b * T_ + tok) * C_;
    const float* tp = t + ((size_t)b * T_ + tok) * C_;
    if (it == 0) __builtin_prefetch(sp + 16 * C_, 0, 0);

    float part[2];
#if HAVE_ASYNC_LDS
    float* lbase = tstage + wave * 2048 + (lane & 15) * 128 + ((lane >> 4) << 6);
#pragma unroll
    for (int j = 0; j < 2; ++j) {
      const v4f* tv = (const v4f*)(tp + (kslot + j) * 64);
      const v4f* sv = (const v4f*)(sp + (kslot + j) * 64);
      // LDS buffer reuse: drain this wave's ds reads before async overwrites
      asm volatile("s_wait_dscnt 0" ::: "memory");
#pragma unroll
      for (int q = 0; q < 16; ++q) {
        __builtin_amdgcn_global_load_async_to_lds_b128(
            (AS_GLOBAL v4i*)(v4i*)(tv + q),
            (AS_LDS v4i*)(v4i*)(lbase + q * 4),
            0, 1 /* CPol TH=NT: stream teacher, keep student in L2 */);
      }
      // student chunk: load + histogram while async copies are in flight
      v4f sreg[16];
#pragma unroll
      for (int q = 0; q < 16; ++q) {
        sreg[q] = sv[q];
#pragma unroll
        for (int c2 = 0; c2 < 4; ++c2) {
          unsigned p = __float_as_uint(sreg[q][c2]) & 0x7fffffffu;
          atomicAdd(&lhist[p >> BINSHIFT], 1u);
        }
      }
      asm volatile("s_wait_asynccnt 0" ::: "memory");
      float acc = 0.0f;
#pragma unroll
      for (int q = 0; q < 16; ++q) {
        v4f tvv = ((const v4f*)lbase)[q];
        v4f d   = sreg[q] - tvv;
        acc += d.x * d.x + d.y * d.y + d.z * d.z + d.w * d.w;
      }
      part[j] = acc;
    }
#else
#pragma unroll
    for (int j = 0; j < 2; ++j) {
      const v4f* sv = (const v4f*)(sp + (kslot + j) * 64);
      const v4f* tv = (const v4f*)(tp + (kslot + j) * 64);
      float acc = 0.0f;
#pragma unroll
      for (int q = 0; q < 16; ++q) {
        v4f a  = sv[q];
        v4f bb = __builtin_nontemporal_load(&tv[q]);   // teacher: stream
        v4f d  = a - bb;
        acc += d.x * d.x + d.y * d.y + d.z * d.z + d.w * d.w;
#pragma unroll
        for (int c2 = 0; c2 < 4; ++c2) {
          unsigned p = __float_as_uint(a[c2]) & 0x7fffffffu;
          atomicAdd(&lhist[p >> BINSHIFT], 1u);
        }
      }
      part[j] = acc;
    }
#endif

    // Reduce 4 K-slot partials per token via matrix pipe: D[m][n] = sum_k A[m][k]
#if WMMA_PATH == 1
    v2f a2 = { part[0], part[1] };
    v2f o2 = { 1.0f, 1.0f };
    v8f cz = {};
    v8f d8 = __builtin_amdgcn_wmma_f32_16x16x4_f32(
        false, a2, false, o2, (short)0, cz, false, false);
#elif WMMA_PATH == 2
    v16h a16 = {};
    a16[0] = (_Float16)part[0];
    a16[1] = (_Float16)part[1];
    v16h o16;
#pragma unroll
    for (int i = 0; i < 16; ++i) o16[i] = (_Float16)1.0f;
    v8f cz = {};
    v8f d8 = __builtin_amdgcn_wmma_f32_16x16x32_f16(
        false, a16, false, o16, (short)0, cz, false, false);
#else
    float tot = part[0] + part[1];
    tot += __shfl_xor(tot, 16, 32);            // pair halves: full token sum
    v8f d8;
#pragma unroll
    for (int r = 0; r < 8; ++r) d8[r] = (r == 0) ? tot : 0.0f;
    dupInv = 1.0f / 2.0f;                      // each token counted twice
#endif

    float rowAcc = 0.0f;
#pragma unroll
    for (int r = 0; r < 8; ++r) {
      float u = d8[r] * (2.0f / (float)C_);    // uncertainty = clip(2*mean)
      u = fminf(fmaxf(u, 0.0f), 1.0f);
      rowAcc += u;
    }
    waveAcc += rowAcc;
  }

  // cross-lane sum (each token's U appears dup times)
  for (int m = 16; m >= 1; m >>= 1) waveAcc += __shfl_xor(waveAcc, m, 32);
  if (lane == 0) atomicAdd(&dsum[b], waveAcc * dupInv);

  __syncthreads();
  for (int i = tid; i < NBIN; i += 256) {
    unsigned v = lhist[i];
    if (v) atomicAdd(&ghist[(size_t)b * NBIN + i], v);
  }
}

// ---------------------------------------------------------------------------
// Kernel 2: per-sample (32 WGs): p_eff, rank k, frac; scan histogram to find
// the bins containing ranks k and k+1 plus the rank offset within candidates.
// ---------------------------------------------------------------------------
__global__ __launch_bounds__(256) void k_select_bins(
    const float* __restrict__ dsum, const float* __restrict__ risk,
    const unsigned* __restrict__ ghist, unsigned* __restrict__ binlo,
    unsigned* __restrict__ binhi, unsigned* __restrict__ r0out,
    float* __restrict__ fracout) {
  const int b = blockIdx.x, tid = threadIdx.x;
  __shared__ unsigned seg[256];
  __shared__ unsigned base[257];
  __shared__ unsigned sh_bk, sh_bk1, sh_start;
  const unsigned* h = ghist + (size_t)b * NBIN;

  unsigned ssum = 0;
  for (int i = 0; i < 32; ++i) ssum += h[tid * 32 + i];
  seg[tid] = ssum;
  __syncthreads();
  if (tid == 0) {
    unsigned run = 0;
    for (int i = 0; i < 256; ++i) { base[i] = run; run += seg[i]; }
    base[256] = run;
  }
  __syncthreads();

  double meanU    = (double)dsum[b] * (1.0 / (double)T_);
  double r        = (double)risk[b];
  double distrust = meanU * (r > 1.0 ? r : 1.0);   // max(U, r*U) summed
  double p        = 0.99 - (0.99 - 0.9) * distrust;
  double pos      = p * (double)(NPS - 1);
  unsigned k      = (unsigned)pos;                  // floor (pos >= 0)
  double frac     = pos - (double)k;

  for (int which = 0; which < 2; ++which) {
    unsigned q = k + which;
    if (base[tid] <= q && q < base[tid + 1]) {
      unsigned cum = base[tid];
      for (int i = 0; i < 32; ++i) {
        unsigned c = h[tid * 32 + i];
        if (q < cum + c) {
          if (which == 0) { sh_bk = (unsigned)(tid * 32 + i); sh_start = cum; }
          else            { sh_bk1 = (unsigned)(tid * 32 + i); }
          break;
        }
        cum += c;
      }
    }
  }
  __syncthreads();
  if (tid == 0) {
    binlo[b]   = sh_bk;
    binhi[b]   = sh_bk1;
    r0out[b]   = k - sh_start;   // rank within candidate set (bins between empty)
    fracout[b] = (float)frac;
  }
}

// ---------------------------------------------------------------------------
// Kernel 3: collect candidates (|s| pattern in target bins). Block-aggregated
// atomics: LDS match buffer, ONE global atomic per WG. s read is L2-hot (RT).
// ---------------------------------------------------------------------------
__global__ __launch_bounds__(256) void k_collect(const float* __restrict__ s,
                                                 const unsigned* __restrict__ binlo,
                                                 const unsigned* __restrict__ binhi,
                                                 unsigned* __restrict__ count,
                                                 unsigned* __restrict__ cand) {
  __shared__ unsigned nMatch;
  __shared__ unsigned sbase;
  __shared__ unsigned mbuf[1024];
  const int tid = threadIdx.x;
  const int b   = blockIdx.x >> 10;   // 1024 WGs per sample (256 thr * 4 floats)
  if (tid == 0) nMatch = 0u;
  __syncthreads();

  const unsigned lo = binlo[b], hi = binhi[b];
  const size_t i4 = (size_t)blockIdx.x * 256 + tid;
  v4f x = ((const v4f*)s)[i4];
#pragma unroll
  for (int c2 = 0; c2 < 4; ++c2) {
    unsigned p   = __float_as_uint(x[c2]) & 0x7fffffffu;
    unsigned bin = p >> BINSHIFT;
    if (bin == lo || bin == hi) {
      unsigned idx = atomicAdd(&nMatch, 1u);
      if (idx < 1024u) mbuf[idx] = p;
    }
  }
  __syncthreads();
  if (tid == 0) {
    unsigned n = nMatch > 1024u ? 1024u : nMatch;
    sbase  = atomicAdd(&count[b], n);
    nMatch = n;
  }
  __syncthreads();
  const unsigned n = nMatch, bs = sbase;
  for (unsigned i = tid; i < n; i += 256) {
    unsigned gi = bs + i;
    if (gi < (unsigned)CAP) cand[(size_t)b * CAP + gi] = mbuf[i];
  }
}

// ---------------------------------------------------------------------------
// Kernel 4: per-sample exact order statistics by binary search on bit patterns
// over the candidate set; linear interpolation -> threshold.
// ---------------------------------------------------------------------------
__global__ __launch_bounds__(256) void k_quantile(
    const unsigned* __restrict__ cand, const unsigned* __restrict__ count,
    const unsigned* __restrict__ binlo, const unsigned* __restrict__ binhi,
    const unsigned* __restrict__ r0v, const float* __restrict__ fracv,
    float* __restrict__ thresh) {
  const int b = blockIdx.x, tid = threadIdx.x;
  __shared__ unsigned red[256];
  const unsigned* c = cand + (size_t)b * CAP;
  unsigned m = count[b]; if (m > (unsigned)CAP) m = (unsigned)CAP;
  const unsigned r0  = r0v[b];
  const unsigned lo0 = binlo[b] << BINSHIFT;
  const unsigned hi0 = ((binhi[b] + 1u) << BINSHIFT) - 1u;

  unsigned v[2];
  for (int which = 0; which < 2; ++which) {
    const unsigned target = r0 + 1u + (unsigned)which;
    unsigned lo = lo0, hi = hi0;
    while (lo < hi) {                       // <= 19 iterations
      unsigned mid = lo + ((hi - lo) >> 1);
      unsigned local = 0;
      for (unsigned i = tid; i < m; i += 256) local += (c[i] <= mid) ? 1u : 0u;
      red[tid] = local;
      __syncthreads();
      for (int st = 128; st > 0; st >>= 1) {
        if (tid < st) red[tid] += red[tid + st];
        __syncthreads();
      }
      unsigned cnt = red[0];
      __syncthreads();
      if (cnt >= target) hi = mid; else lo = mid + 1u;
    }
    v[which] = lo;
  }
  if (tid == 0) {
    float f0 = __uint_as_float(v[0]);
    float f1 = __uint_as_float(v[1]);
    thresh[b] = f0 + fracv[b] * (f1 - f0);
  }
}

// ---------------------------------------------------------------------------
// Kernel 5: out = clip(s, -t, t). Last use of s -> NT load; NT store.
// ---------------------------------------------------------------------------
__global__ __launch_bounds__(256) void k_clip(const float* __restrict__ s,
                                              const float* __restrict__ thresh,
                                              float* __restrict__ out) {
  const size_t i4 = (size_t)blockIdx.x * 256 + threadIdx.x;
  const int b = (int)(i4 >> 18);              // 2^18 float4 per sample
  const float tt = thresh[b];
  v4f x = __builtin_nontemporal_load(&((const v4f*)s)[i4]);
  v4f y;
  y.x = fminf(fmaxf(x.x, -tt), tt);
  y.y = fminf(fmaxf(x.y, -tt), tt);
  y.z = fminf(fmaxf(x.z, -tt), tt);
  y.w = fminf(fmaxf(x.w, -tt), tt);
  __builtin_nontemporal_store(y, &((v4f*)out)[i4]);
}

extern "C" void kernel_launch(void* const* d_in, const int* in_sizes, int n_in,
                              void* d_out, int out_size, void* d_ws, size_t ws_size,
                              hipStream_t stream) {
  const float* s    = (const float*)d_in[0];
  const float* t    = (const float*)d_in[1];
  const float* risk = (const float*)d_in[2];
  float* out = (float*)d_out;

  char* ws = (char*)d_ws;
  float*    dsum   = (float*)(ws + OFF_DSUM);
  unsigned* count  = (unsigned*)(ws + OFF_COUNT);
  unsigned* binlo  = (unsigned*)(ws + OFF_BINLO);
  unsigned* binhi  = (unsigned*)(ws + OFF_BINHI);
  unsigned* r0v    = (unsigned*)(ws + OFF_R0);
  float*    fracv  = (float*)(ws + OFF_FRAC);
  float*    thresh = (float*)(ws + OFF_THRESH);
  unsigned* ghist  = (unsigned*)(ws + OFF_HIST);
  unsigned* cand   = (unsigned*)(ws + OFF_CAND);

  (void)hipMemsetAsync(ws, 0, ZERO_BYTES, stream);

  k_pass_a<<<B_ * 16, 256, 0, stream>>>(s, t, dsum, ghist);
  k_select_bins<<<B_, 256, 0, stream>>>(dsum, risk, ghist, binlo, binhi, r0v, fracv);
  k_collect<<<B_ * 1024, 256, 0, stream>>>(s, binlo, binhi, count, cand);
  k_quantile<<<B_, 256, 0, stream>>>(cand, count, binlo, binhi, r0v, fracv, thresh);
  k_clip<<<(B_ * NPS) / (256 * 4), 256, 0, stream>>>(s, thresh, out);
}
